// ACTimeAwareMultiHeadAttention_88450556494376
// MI455X (gfx1250) — compile-verified, matrix operand order checked by reading, not run on hardware
//
#include <hip/hip_runtime.h>
#include <hip/hip_bf16.h>

#define B_  8
#define S_  200
#define H_  4
#define D_  64
#define HID 256
#define BS  (B_ * S_)      // 1600
#define S_PAD 208

typedef _Float16 v16h __attribute__((ext_vector_type(16)));
typedef float    v8f  __attribute__((ext_vector_type(8)));

// ---------------------------------------------------------------------------
// Wave32 / block reduction helpers
// ---------------------------------------------------------------------------
__device__ __forceinline__ float wave_sum(float v) {
#pragma unroll
    for (int o = 16; o >= 1; o >>= 1) v += __shfl_xor(v, o, 32);
    return v;
}

__device__ __forceinline__ float4 wave_sum4(float4 v) {
#pragma unroll
    for (int o = 16; o >= 1; o >>= 1) {
        v.x += __shfl_xor(v.x, o, 32);
        v.y += __shfl_xor(v.y, o, 32);
        v.z += __shfl_xor(v.z, o, 32);
        v.w += __shfl_xor(v.w, o, 32);
    }
    return v;
}

__device__ __forceinline__ float4 wave_max4(float4 v) {
#pragma unroll
    for (int o = 16; o >= 1; o >>= 1) {
        v.x = fmaxf(v.x, __shfl_xor(v.x, o, 32));
        v.y = fmaxf(v.y, __shfl_xor(v.y, o, 32));
        v.z = fmaxf(v.z, __shfl_xor(v.z, o, 32));
        v.w = fmaxf(v.w, __shfl_xor(v.w, o, 32));
    }
    return v;
}

// 256-thread (8-wave) block reductions; red has >= 9 slots.
__device__ __forceinline__ float block_sum(float v, float* red, int t) {
    v = wave_sum(v);
    const int wid = t >> 5, lane = t & 31;
    if (lane == 0) red[wid] = v;
    __syncthreads();
    if (wid == 0) {
        float w = (lane < 8) ? red[lane] : 0.f;
        w = wave_sum(w);
        if (lane == 0) red[8] = w;
    }
    __syncthreads();
    float r = red[8];
    __syncthreads();
    return r;
}

__device__ __forceinline__ float4 block_sum4(float4 v, float4* red, int t) {
    v = wave_sum4(v);
    const int wid = t >> 5, lane = t & 31;
    if (lane == 0) red[wid] = v;
    __syncthreads();
    if (wid == 0) {
        float4 w = (lane < 8) ? red[lane] : make_float4(0.f, 0.f, 0.f, 0.f);
        w = wave_sum4(w);
        if (lane == 0) red[8] = w;
    }
    __syncthreads();
    float4 r = red[8];
    __syncthreads();
    return r;
}

__device__ __forceinline__ float4 block_max4(float4 v, float4* red, int t) {
    v = wave_max4(v);
    const int wid = t >> 5, lane = t & 31;
    if (lane == 0) red[wid] = v;
    __syncthreads();
    if (wid == 0) {
        float4 w = (lane < 8) ? red[lane]
                              : make_float4(-INFINITY, -INFINITY, -INFINITY, -INFINITY);
        w = wave_max4(w);
        if (lane == 0) red[8] = w;
    }
    __syncthreads();
    float4 r = red[8];
    __syncthreads();
    return r;
}

// ---------------------------------------------------------------------------
// Kernel 1: Q/K/V projection via v_wmma_f32_16x16x32_f16.
// One wave per 16x16 output tile; K-loop of 8 WMMAs (K=256).
// A/B fragment element -> (row/col, k) mapping per CDNA5 ISA 16-bit layout:
//   k(e, lane) = ((e>>3)<<4) + ((lane>>4)<<3) + (e&7)
// ---------------------------------------------------------------------------
__device__ __forceinline__ int wmma_k_of(int e, int lane) {
    return ((e >> 3) << 4) + ((lane >> 4) << 3) + (e & 7);
}

__global__ void qkv_gemm_wmma(const float* __restrict__ X,
                              const float* __restrict__ Wq, const float* __restrict__ bq,
                              const float* __restrict__ Wk, const float* __restrict__ bk,
                              const float* __restrict__ Wv, const float* __restrict__ bv,
                              float* __restrict__ qbuf, float* __restrict__ kbuf,
                              float* __restrict__ vbuf) {
    const int lane = threadIdx.x & 31;
    const int wave = threadIdx.x >> 5;
    const int tile = blockIdx.x * 8 + wave;   // 0..4799
    const int w    = tile / (BS);             // which weight (1600 tiles each)
    const int t2   = tile % (BS);
    const int mt   = t2 >> 4;                 // 0..99  (M tiles: 1600/16)
    const int nt   = t2 & 15;                 // 0..15  (N tiles: 256/16)

    const float* W;
    const float* bias;
    float* outp;
    if (w == 0)      { W = Wq; bias = bq; outp = qbuf; }
    else if (w == 1) { W = Wk; bias = bk; outp = kbuf; }
    else             { W = Wv; bias = bv; outp = vbuf; }

    const int m0 = mt * 16, n0 = nt * 16;
    const float* Xrow = X + (long)(m0 + (lane & 15)) * HID;
    const int ncol = n0 + (lane & 15);

    v8f acc = {};
#pragma unroll
    for (int kb = 0; kb < HID; kb += 32) {
        v16h af, bf;
#pragma unroll
        for (int e = 0; e < 16; ++e) {
            const int kk = kb + wmma_k_of(e, lane);
            af[e] = (_Float16)Xrow[kk];
            bf[e] = (_Float16)W[kk * HID + ncol];
        }
        acc = __builtin_amdgcn_wmma_f32_16x16x32_f16(false, af, false, bf,
                                                     (short)0, acc, false, false);
    }

    const float bval = bias[ncol];
#pragma unroll
    for (int r = 0; r < 8; ++r) {
        const int m = m0 + ((lane >> 4) << 3) + r;
        outp[(long)m * HID + ncol] = acc[r] + bval;
    }
}

// ---------------------------------------------------------------------------
// Kernel 2: fold posK into k, posV into v (in place), and compute per-(row,head)
// scalars {qo, qd, ko, kd} with wave32 reductions.
// ---------------------------------------------------------------------------
__global__ void prep_kernel(const float* __restrict__ posK, const float* __restrict__ posV,
                            const float* __restrict__ order_w, const float* __restrict__ dist_w,
                            const float* __restrict__ qbuf,
                            float* __restrict__ kbuf, float* __restrict__ vbuf,
                            float4* __restrict__ scal) {
    __shared__ float4 red4[8];
    const int r = blockIdx.x;       // 0..BS-1
    const int t = threadIdx.x;      // 0..255
    const int dh = t & 63;
    const long idx = (long)r * HID + t;

    const float qv = qbuf[idx];
    const float kv = kbuf[idx];
    const float vv = vbuf[idx];
    kbuf[idx] = kv + posK[idx];
    vbuf[idx] = vv + posV[idx];

    float4 p;
    p.x = qv * order_w[dh];
    p.y = qv * dist_w[dh];
    p.z = kv * order_w[64 + dh];
    p.w = kv * dist_w[64 + dh];
    p = wave_sum4(p);

    const int wid = t >> 5, lane = t & 31;
    if (lane == 0) red4[wid] = p;
    __syncthreads();
    if (t < 4) {   // head h = t spans waves 2t and 2t+1
        const float4 a = red4[2 * t], b = red4[2 * t + 1];
        scal[r * 4 + t] = make_float4(a.x + b.x, a.y + b.y, a.z + b.z, a.w + b.w);
    }
}

// ---------------------------------------------------------------------------
// Kernel 3: fused attention. One block per (b,i); streams timeK/timeV once.
// ---------------------------------------------------------------------------
__global__ void attn_kernel(const float* __restrict__ qbuf,
                            const float* __restrict__ kbuf,   // k + posK
                            const float* __restrict__ vbuf,   // v + posV
                            const float4* __restrict__ scal,
                            const float* __restrict__ timeK,
                            const float* __restrict__ timeV,
                            const float* __restrict__ mask,
                            const float* __restrict__ order_b,
                            const float* __restrict__ dist_b,
                            const float* __restrict__ scalar,
                            float* __restrict__ ctxf) {
    __shared__ __align__(16) float q_sh[HID];
    __shared__ float probs_sh[4][S_PAD];
    __shared__ float4 red4[9];
    __shared__ float qoi_sh[4], qdi_sh[4];

    const int blk = blockIdx.x;
    const int b = blk / S_, i = blk % S_;
    const int t = threadIdx.x;

    q_sh[t] = qbuf[(long)(b * S_ + i) * HID + t];
    if (t < 4) {
        const float4 sq = scal[(b * S_ + i) * 4 + t];
        qoi_sh[t] = sq.x;
        qdi_sh[t] = sq.y;
    }
    const float ob   = order_b[0];
    const float db   = dist_b[0];
    const float sc   = scalar[0];
    const float sc2h = 0.5f * sc * sc;
    __syncthreads();

    // ---- Phase 1: scores for j = t (4 heads) -------------------------------
    const int j = t;
    float4 s4 = make_float4(-INFINITY, -INFINITY, -INFINITY, -INFINITY);
    if (j < S_) {
        const float4* q4 = reinterpret_cast<const float4*>(q_sh);
        const float4* kr = reinterpret_cast<const float4*>(kbuf + (long)(b * S_ + j) * HID);
        const float4* tk = reinterpret_cast<const float4*>(
            timeK + ((long)(b * S_ + i) * S_ + j) * HID);

        float acc[4];
#pragma unroll
        for (int h = 0; h < 4; ++h) {
            float a = 0.f;
#pragma unroll
            for (int d4 = 0; d4 < 16; ++d4) {
                const float4 qv = q4[h * 16 + d4];
                const float4 kv = kr[h * 16 + d4];
                const float4 tv = tk[h * 16 + d4];
                a += qv.x * (kv.x + tv.x);
                a += qv.y * (kv.y + tv.y);
                a += qv.z * (kv.z + tv.z);
                a += qv.w * (kv.w + tv.w);
            }
            acc[h] = a;
        }

        const float mval  = mask[((long)b * S_ + i) * S_ + j];
        const float gdist = logf((float)(i > j ? i - j : j - i) + 1.0f);
        float sres[4];
#pragma unroll
        for (int h = 0; h < 4; ++h) {
            const float4 sj = scal[(b * S_ + j) * 4 + h];  // {qo_j, qd_j, ko_j, kd_j}
            // order term: log sigmoid(x) = -softplus(-x); log(1-sigmoid) = -softplus(x)
            const float x  = qoi_sh[h] + sj.z + ob;
            const float sp = log1pf(expf(-fabsf(x)));
            float eo = (j > i) ? -(sp + fmaxf(-x, 0.f))
                               : -(sp + fmaxf(x, 0.f));
            eo = fmaxf(eo, -55.262043f);               // log(1e-24) floor
            // distance term
            const float pd  = qdi_sh[h] + sj.w + db;
            const float dgd = gdist - pd;
            const float ed  = -dgd * dgd * sc2h;
            sres[h] = (acc[h] + eo + ed) * 0.125f + mval;   // /sqrt(64) then +mask
        }
        s4 = make_float4(sres[0], sres[1], sres[2], sres[3]);
    }

    // ---- Phase 2: softmax over j (all 4 heads at once) ---------------------
    const float4 mx = block_max4(s4, red4, t);
    float4 p = make_float4(0.f, 0.f, 0.f, 0.f);
    if (j < S_) {
        p.x = expf(s4.x - mx.x);
        p.y = expf(s4.y - mx.y);
        p.z = expf(s4.z - mx.z);
        p.w = expf(s4.w - mx.w);
    }
    const float4 sum = block_sum4(p, red4, t);
    if (j < S_) {
        probs_sh[0][j] = p.x / sum.x;
        probs_sh[1][j] = p.y / sum.y;
        probs_sh[2][j] = p.z / sum.z;
        probs_sh[3][j] = p.w / sum.w;
    }
    __syncthreads();

    // ---- Phase 3: context; thread t -> column (h = t/64, d = t%64) ---------
    const int h = t >> 6;
    const float* vrow = vbuf + (long)b * S_ * HID + t;
    const float* tvr  = timeV + (long)(b * S_ + i) * S_ * HID + t;
    float acc = 0.f;
#pragma unroll 4
    for (int jj = 0; jj < S_; ++jj) {
        acc += probs_sh[h][jj] * (vrow[(long)jj * HID] + tvr[(long)jj * HID]);
    }
    ctxf[(long)(b * S_ + i) * HID + t] = acc;
}

// ---------------------------------------------------------------------------
// Kernel 4: output projection + residual + LayerNorm. One block per row.
// ---------------------------------------------------------------------------
__global__ void outproj_ln(const float* __restrict__ ctxf,
                           const float* __restrict__ Wd, const float* __restrict__ bd,
                           const float* __restrict__ x_in,
                           const float* __restrict__ ln_g, const float* __restrict__ ln_b,
                           float* __restrict__ out) {
    __shared__ float row_sh[HID];
    __shared__ float red[9];
    const int r = blockIdx.x;
    const int t = threadIdx.x;

    row_sh[t] = ctxf[(long)r * HID + t];
    __syncthreads();

    float acc = bd[t];
#pragma unroll 8
    for (int k = 0; k < HID; ++k) acc += row_sh[k] * Wd[k * HID + t];

    const float hs = acc + x_in[(long)r * HID + t];
    const float mu = block_sum(hs, red, t) * (1.0f / HID);
    const float dd = hs - mu;
    const float var = block_sum(dd * dd, red, t) * (1.0f / HID);
    out[(long)r * HID + t] = dd * rsqrtf(var + 1e-12f) * ln_g[t] + ln_b[t];
}

// ---------------------------------------------------------------------------
// Launch
// ---------------------------------------------------------------------------
extern "C" void kernel_launch(void* const* d_in, const int* in_sizes, int n_in,
                              void* d_out, int out_size, void* d_ws, size_t ws_size,
                              hipStream_t stream) {
    const float* X     = (const float*)d_in[0];
    const float* mask  = (const float*)d_in[1];
    const float* posK  = (const float*)d_in[2];
    const float* posV  = (const float*)d_in[3];
    const float* timeK = (const float*)d_in[4];
    const float* timeV = (const float*)d_in[5];
    const float* Wq = (const float*)d_in[6];
    const float* bq = (const float*)d_in[7];
    const float* Wk = (const float*)d_in[8];
    const float* bk = (const float*)d_in[9];
    const float* Wv = (const float*)d_in[10];
    const float* bv = (const float*)d_in[11];
    const float* order_w = (const float*)d_in[12];
    const float* order_b = (const float*)d_in[13];
    const float* dist_w  = (const float*)d_in[14];
    const float* dist_b  = (const float*)d_in[15];
    const float* scalar  = (const float*)d_in[16];
    const float* Wd = (const float*)d_in[17];
    const float* bd = (const float*)d_in[18];
    const float* ln_g = (const float*)d_in[19];
    const float* ln_b = (const float*)d_in[20];
    float* out = (float*)d_out;

    float* ws = (float*)d_ws;
    float*  qbuf = ws;                      // [BS, 256]
    float*  kbuf = ws + 409600;             // [BS, 256] -> k + posK
    float*  vbuf = ws + 819200;             // [BS, 256] -> v + posV
    float*  ctxf = ws + 1228800;            // [BS, 256]
    float4* scal = (float4*)(ws + 1638400); // [BS, 4] of {qo,qd,ko,kd}

    qkv_gemm_wmma<<<600, 256, 0, stream>>>(X, Wq, bq, Wk, bk, Wv, bv,
                                           qbuf, kbuf, vbuf);
    prep_kernel<<<BS, 256, 0, stream>>>(posK, posV, order_w, dist_w,
                                        qbuf, kbuf, vbuf, scal);
    attn_kernel<<<BS, 256, 0, stream>>>(qbuf, kbuf, vbuf, scal, timeK, timeV,
                                        mask, order_b, dist_b, scalar, ctxf);
    outproj_ln<<<BS, 256, 0, stream>>>(ctxf, Wd, bd, X, ln_g, ln_b, out);
}
